// PolarSimReg_24584392802995
// MI455X (gfx1250) — matrix operand order repaired and checked by reading, not compiled, not run on hardware
//
#include <hip/hip_runtime.h>
#include <hip/hip_bf16.h>

typedef __attribute__((ext_vector_type(16))) _Float16 v16h;
typedef __attribute__((ext_vector_type(8)))  _Float16 v8h;
typedef __attribute__((ext_vector_type(8)))  float    v8f;

#define NROWS 8192
#define KDIM  256
#define INVT  (1.0f / 0.7f)
#define MARG  0.5f
#define NBT   (NROWS / 128)              /* 64 macro-tiles per dim */
#define NTRI  (NBT * (NBT + 1) / 2)      /* 2080 upper-tri blocks  */
#define BROWS 128                        /* B panel rows per block */
#define BSTR  264                        /* padded LDS row stride (f16): 132 dw = 4 mod 64 banks */

union V16 { v16h v; v8h h[2]; };

// ---------------------------------------------------------------------------
// 0) zero accumulators (must run every launch; graph replays don't re-poison)
// ---------------------------------------------------------------------------
__global__ void init_kernel(float* __restrict__ A, float* __restrict__ B,
                            int* __restrict__ hist, float* __restrict__ out) {
  int i = blockIdx.x * blockDim.x + threadIdx.x;
  if (i < NROWS) { A[i] = 0.0f; B[i] = 0.0f; }
  if (i < 16)    hist[i] = 0;
  if (i == 0)    *out = 0.0f;
}

// ---------------------------------------------------------------------------
// 1) L2-normalize rows, emit f16 (WMMA operand precision)
// ---------------------------------------------------------------------------
__global__ __launch_bounds__(256)
void normalize_kernel(const float* __restrict__ proj, _Float16* __restrict__ Pn) {
  const int row = blockIdx.x;
  const int tid = threadIdx.x;
  float v  = proj[row * KDIM + tid];
  float ss = v * v;
  #pragma unroll
  for (int off = 16; off > 0; off >>= 1) ss += __shfl_xor(ss, off, 32);

  __shared__ float red[8];
  __shared__ float sscale;
  const int wave = tid >> 5, lane = tid & 31;
  if (lane == 0) red[wave] = ss;
  __syncthreads();
  if (tid == 0) {
    float t = 0.0f;
    #pragma unroll
    for (int w = 0; w < 8; ++w) t += red[w];
    sscale = rsqrtf(t);
  }
  __syncthreads();
  Pn[row * KDIM + tid] = (_Float16)(v * sscale);
}

// ---------------------------------------------------------------------------
// 2) class histogram (for the empty-set -> 1.0 substitution)
// ---------------------------------------------------------------------------
__global__ void hist_kernel(const long long* __restrict__ y, int* __restrict__ hist) {
  int i = blockIdx.x * blockDim.x + threadIdx.x;
  if (i < NROWS) atomicAdd(&hist[(int)y[i]], 1);
}

// ---------------------------------------------------------------------------
// 3) fused sim-tile over the UPPER TRIANGLE of the symmetric sim matrix.
//    block = 8 waves = 128x128 tile; wave = 16x128 strip; 8 WMMA col tiles.
//    The 128x256 f16 B panel is async-copied to LDS once per block
//    (GLOBAL_LOAD_ASYNC_TO_LDS_B128, ASYNCcnt) and shared by all 8 waves.
// ---------------------------------------------------------------------------
__global__ __launch_bounds__(256)
void simtile_kernel(const _Float16* __restrict__ Pn, const long long* __restrict__ y,
                    float* __restrict__ A, float* __restrict__ B) {
  // linear block index -> (by, bx) with bx >= by (triangular inversion)
  const int t  = blockIdx.x;
  int bx = (int)((sqrtf(8.0f * (float)t + 1.0f) - 1.0f) * 0.5f);
  while ((bx + 1) * (bx + 2) / 2 <= t) ++bx;
  while (bx * (bx + 1) / 2 > t)        --bx;
  const int by = t - bx * (bx + 1) / 2;
  const bool offdiag = (bx > by);      // uniform over the block

  const int tid  = threadIdx.x;
  const int wave = tid >> 5;
  const int lane = tid & 31;
  const int m    = lane & 15;          // column within 16-wide tile (C-frag N)
  const int sel  = lane >> 4;          // half-wave select

  const int i0 = by * 128 + wave * 16; // row strip for this wave
  const int j0 = bx * 128;             // 128 columns for the block

  __shared__ _Float16 panelB[BROWS * BSTR];   // 67.6 KB, padded rows
  __shared__ float colA_s[128];
  __shared__ float colB_s[128];
  if (tid < 128) { colA_s[tid] = 0.0f; colB_s[tid] = 0.0f; }

  // --- async-copy B panel (rows j0..j0+127, K 0..255) into LDS -------------
  // 4096 16-byte chunks, 16 per lane. LDS dest offset = low 32 bits of the
  // generic shared pointer (flat->LDS keeps the LDS byte offset in addr[31:0]).
  {
    #pragma unroll
    for (int it = 0; it < 16; ++it) {
      const int c   = tid + 256 * it;       // chunk id
      const int row = c >> 5;               // 0..127
      const int kc  = (c & 31) * 8;         // 0..248, step 8 f16 (16 B)
      const unsigned laddr =
          (unsigned)(uintptr_t)(&panelB[row * BSTR + kc]);
      const unsigned long long gaddr =
          (unsigned long long)(uintptr_t)(Pn + (size_t)(j0 + row) * KDIM + kc);
      asm volatile("global_load_async_to_lds_b128 %0, %1, off"
                   :: "v"(laddr), "v"(gaddr) : "memory");
    }
    asm volatile("s_wait_asynccnt 0x0" ::: "memory");
  }
  __syncthreads();   // panel (and colA_s/colB_s init) visible to all waves

  v8f acc[8] = {};                     // 8 column tiles of 16x16 f32

  // A-frag (global): lane holds row i0+m; lanes 0-15 K {kb..kb+7, kb+16..23},
  //                  lanes 16-31 K {kb+8..15, kb+24..31}  (ISA 16-bit A layout)
  // B-frag (LDS): lane holds row jt*16+m; K {kb+sel*8..+7, kb+16+sel*8..+7} (same as A)
  const _Float16* arow  = Pn + (size_t)(i0 + m) * KDIM;
  const _Float16* browl = &panelB[m * BSTR];

  for (int kb = 0; kb < KDIM; kb += 32) {
    V16 a;
    const _Float16* ap = arow + kb + sel * 8;
    a.h[0] = *(const v8h*)(ap);
    a.h[1] = *(const v8h*)(ap + 16);
    #pragma unroll
    for (int jt = 0; jt < 8; ++jt) {
      V16 b;
      const _Float16* bp = browl + (size_t)jt * 16 * BSTR + kb + sel * 8;
      b.h[0] = *(const v8h*)(bp);          // ds_load_b128
      b.h[1] = *(const v8h*)(bp + 16);     // ds_load_b128
      acc[jt] = __builtin_amdgcn_wmma_f32_16x16x32_f16(
          false, a.v, false, b.v, (short)0, acc[jt], false, false);
    }
  }

  // labels for the 8 columns this lane owns (independent of r)
  long long yj[8];
  #pragma unroll
  for (int jt = 0; jt < 8; ++jt) yj[jt] = y[j0 + jt * 16 + m];

  float colAp[8] = {0,0,0,0,0,0,0,0};  // per-lane column partials (off-diag)
  float colBp[8] = {0,0,0,0,0,0,0,0};

  // C-frag: VGPR r holds (row = r + sel*8, col = m)
  #pragma unroll
  for (int r = 0; r < 8; ++r) {
    const int i = i0 + sel * 8 + r;
    const long long yi = y[i];
    float sA = 0.0f, sB = 0.0f;
    #pragma unroll
    for (int jt = 0; jt < 8; ++jt) {
      const int j = j0 + jt * 16 + m;
      const float s = acc[jt][r];
      if (yi != yj[jt]) {                       // different class -> A
        const float e = __expf(s * INVT);
        sA += e;
        if (offdiag) colAp[jt] += e;            // symmetric: also A[j]
      } else if (i != j) {                      // same class, not self -> B
        const float e = __expf(-s * INVT);
        sB += e;
        if (offdiag) colBp[jt] += e;            // symmetric: also B[j]
      }
    }
    // row i lives on 16 lanes of this half-wave: reduce across width 16
    #pragma unroll
    for (int off = 8; off > 0; off >>= 1) {
      sA += __shfl_xor(sA, off, 16);
      sB += __shfl_xor(sB, off, 16);
    }
    if (m == 0) {
      atomicAdd(&A[i], sA);
      atomicAdd(&B[i], sB);
    }
  }

  // column-direction accumulation (only strictly-upper blocks; every i < j)
  if (offdiag) {
    #pragma unroll
    for (int jt = 0; jt < 8; ++jt) {
      atomicAdd(&colA_s[jt * 16 + m], colAp[jt]);   // ds_add_f32
      atomicAdd(&colB_s[jt * 16 + m], colBp[jt]);
    }
    __syncthreads();
    if (tid < 128) {
      atomicAdd(&A[j0 + tid], colA_s[tid]);
      atomicAdd(&B[j0 + tid], colB_s[tid]);
    }
  }
}

// ---------------------------------------------------------------------------
// 4) per-row loss + mean reduction
// ---------------------------------------------------------------------------
__global__ __launch_bounds__(256)
void finalize_kernel(const float* __restrict__ A, const float* __restrict__ B,
                     const long long* __restrict__ y, const int* __restrict__ hist,
                     float* __restrict__ out) {
  const int i = blockIdx.x * blockDim.x + threadIdx.x;
  float li = 0.0f;
  if (i < NROWS) {
    const int c = hist[(int)y[i]];
    const float a = (NROWS - c > 0) ? A[i] : 1.0f;  // no diff-class -> exp(0)=1
    const float b = (c - 1    > 0) ? B[i] : 1.0f;   // no same-class -> exp(0)=1
    li = log1pf(__expf(MARG * INVT) * a * b);
  }
  #pragma unroll
  for (int off = 16; off > 0; off >>= 1) li += __shfl_xor(li, off, 32);
  __shared__ float red[8];
  const int wave = threadIdx.x >> 5, lane = threadIdx.x & 31;
  if (lane == 0) red[wave] = li;
  __syncthreads();
  if (threadIdx.x == 0) {
    float t = 0.0f;
    #pragma unroll
    for (int w = 0; w < 8; ++w) t += red[w];
    atomicAdd(out, t * (1.0f / NROWS));
  }
}

// ---------------------------------------------------------------------------
extern "C" void kernel_launch(void* const* d_in, const int* in_sizes, int n_in,
                              void* d_out, int out_size, void* d_ws, size_t ws_size,
                              hipStream_t stream) {
  const float*     proj = (const float*)d_in[0];
  const long long* y    = (const long long*)d_in[1];

  _Float16* Pn   = (_Float16*)d_ws;                                  // 4 MB
  float*    A    = (float*)((char*)d_ws + (size_t)NROWS * KDIM * 2); // 32 KB
  float*    B    = A + NROWS;                                        // 32 KB
  int*      hist = (int*)(B + NROWS);                                // 64 B
  float*    out  = (float*)d_out;

  init_kernel     <<<dim3(NROWS / 256), 256, 0, stream>>>(A, B, hist, out);
  normalize_kernel<<<dim3(NROWS),       256, 0, stream>>>(proj, Pn);
  hist_kernel     <<<dim3(NROWS / 256), 256, 0, stream>>>(y, hist);
  simtile_kernel  <<<dim3(NTRI),        256, 0, stream>>>(Pn, y, A, B);
  finalize_kernel <<<dim3(NROWS / 256), 256, 0, stream>>>(A, B, y, hist, out);
}
